// TargetAttentionLayer_23579370455264
// MI455X (gfx1250) — matrix operand order repaired and checked by reading, not compiled, not run on hardware
//
#include <hip/hip_runtime.h>

typedef __attribute__((ext_vector_type(16))) __bf16 v16bf;
typedef __attribute__((ext_vector_type(8)))  float  v8f;
typedef __attribute__((ext_vector_type(4)))  unsigned int u32x4;
typedef __attribute__((ext_vector_type(8)))  int    i32x8;
typedef __attribute__((ext_vector_type(4)))  int    i32x4;

#define BATCH 2048
#define SEQ   200
#define DIM   128
#define HID   64
#define SPAD  208   // SEQ padded to 13 WMMA M-tiles
#define MTILES 13
#define NEG_BIG (-1e30f)

__global__ __launch_bounds__(256)
void target_attn_kernel(const float* __restrict__ his,
                        const float* __restrict__ target,
                        const float* __restrict__ mask,
                        const float* __restrict__ W,
                        const float* __restrict__ Wb,
                        const float* __restrict__ O,
                        float* __restrict__ out)
{
    __shared__ float  his_s[SPAD * DIM];        // 106496 B, f32 (GEMM A source + pooling)
    __shared__ __bf16 w1_s[DIM * HID];          // 16384 B, fragment-major bf16
    __shared__ float  c_s[HID];                 // per-batch constant (tgt @ (Wt-Wd) + bias)
    __shared__ float  o_s[HID];
    __shared__ float  alpha_s[SPAD];
    __shared__ float  red_s[16];
    __shared__ float  pool_s[DIM];

    const int b    = blockIdx.x;
    const int tid  = threadIdx.x;
    const int wave = tid >> 5;
    const int lane = tid & 31;

    // ---------- Phase 0: stage inputs ----------
    // his[b]: one TDM DMA of the contiguous 200*128 f32 tile straight into LDS.
    if (wave == 0) {
        const unsigned long long ga =
            (unsigned long long)(const void*)(his + (size_t)b * SEQ * DIM);
        const unsigned int laddr = (unsigned int)(size_t)(void*)his_s; // low 32 = LDS byte addr

        // D# group 0: count=1 | lds_addr | global_addr[56:0] | type=2
        u32x4 g0;
        g0.x = 0x1u;                                    // count=1, user mode, no gather
        g0.y = laddr;                                   // lds_addr (bits 63:32)
        g0.z = (unsigned int)ga;                        // global_addr[31:0]
        g0.w = (unsigned int)((ga >> 32) & 0x1FFFFFFu)  // global_addr[56:32]
             | (2u << 30);                              // type=2 ("image")

        // D# group 1: 1-D tile of 25600 4-byte elements
        const unsigned int NELEM = SEQ * DIM;           // 25600
        i32x8 g1;
        g1[0] = (int)(2u << 16);                        // data_size=2 (4B); wg_mask=0
        g1[1] = (int)((NELEM & 0xFFFFu) << 16);         // tensor_dim0[15:0] @ bits 63:48
        g1[2] = (int)((NELEM >> 16) & 0xFFFFu);         // tensor_dim0[31:16]; tensor_dim1=0
        g1[3] = (int)((NELEM & 0xFFFFu) << 16);         // tile_dim0 @ bits 127:112
        g1[4] = 0;                                      // tile_dim1=0, tile_dim2=0 (1-D)
        g1[5] = (int)NELEM;                             // tensor_dim0_stride[31:0]
        g1[6] = 0;                                      // stride hi / dim1_stride lo
        g1[7] = 0;

        i32x4 z4 = {0, 0, 0, 0};
        i32x8 z8 = {0, 0, 0, 0, 0, 0, 0, 0};
        __builtin_amdgcn_tensor_load_to_lds(g0, g1, z4, z4, z8, 0);
    }

    // Meanwhile: other waves (and wave 0 lanes) prep small LDS state.
    // zero the 8 padded rows so WMMA on them stays finite (disjoint from TDM region)
    #pragma unroll
    for (int i = 0; i < (8 * DIM) / 256; ++i)
        his_s[SEQ * DIM + tid + i * 256] = 0.f;

    // W1 = W[0:128] + W[128:256], bf16, stored fragment-major:
    // element (K,n) -> ((nt*4+kt)*32 + hi*16 + laneN)*16 + j
    for (int i = tid; i < DIM * HID; i += 256) {
        int K = i >> 6, n = i & 63;
        float v = W[K * HID + n] + W[(DIM + K) * HID + n];
        int nt = n >> 4, ln = n & 15;
        int kt = K >> 5, hi2 = (K >> 4) & 1, j = K & 15;
        w1_s[((nt * 4 + kt) * 32 + hi2 * 16 + ln) * 16 + j] = (__bf16)v;
    }
    // c = Wb + tgt @ (W[256:384] - W[128:256]) ; o = O
    if (tid < HID) {
        const float* tg = target + (size_t)b * DIM;
        float acc = Wb[tid];
        #pragma unroll 4
        for (int d = 0; d < DIM; ++d)
            acc = __builtin_fmaf(tg[d], W[(2 * DIM + d) * HID + tid] - W[(DIM + d) * HID + tid], acc);
        c_s[tid] = acc;
        o_s[tid] = O[tid];
    }

    if (wave == 0)
        __builtin_amdgcn_s_wait_tensorcnt(0);   // wave 0 owns the TENSORcnt
    __syncthreads();                            // publish LDS to all waves

    // ---------- Phase 1: WMMA GEMM -> alpha ----------
    const int laneN = lane & 15;
    const int hi    = lane >> 4;

    // all 16 B fragments resident in VGPRs (32B contiguous LDS load each)
    v16bf bfrag[4][4];
    {
        const v16bf* wp = (const v16bf*)w1_s;
        #pragma unroll
        for (int nt = 0; nt < 4; ++nt)
            #pragma unroll
            for (int kt = 0; kt < 4; ++kt)
                bfrag[nt][kt] = wp[(nt * 4 + kt) * 32 + lane];
    }

    for (int m = wave; m < MTILES; m += 8) {
        const int row = m * 16 + laneN;
        // A fragments: contiguous k order (same as B in w1_s), K = kt*32 + hi*16 + j
        v16bf afrag[4];
        #pragma unroll
        for (int kt = 0; kt < 4; ++kt)
            #pragma unroll
            for (int j = 0; j < 16; ++j) {
                int K = kt * 32 + hi * 16 + j;
                afrag[kt][j] = (__bf16)his_s[row * DIM + K];
            }

        float p[8] = {0.f, 0.f, 0.f, 0.f, 0.f, 0.f, 0.f, 0.f};
        #pragma unroll
        for (int nt = 0; nt < 4; ++nt) {
            v8f acc = {};
            #pragma unroll
            for (int kt = 0; kt < 4; ++kt)
                acc = __builtin_amdgcn_wmma_f32_16x16x32_bf16(
                        false, afrag[kt], false, bfrag[nt][kt],
                        (short)0, acc, false, false);
            // C layout: vgpr r, lanes 0-15 -> M=r, N=lane; lanes 16-31 -> M=8+r
            float cc = c_s[nt * 16 + laneN];
            float oc = o_s[nt * 16 + laneN];
            #pragma unroll
            for (int r = 0; r < 8; ++r) {
                float h = fmaxf(acc[r] + cc, 0.f);     // relu(his@W1 + c)
                p[r] = __builtin_fmaf(h, oc, p[r]);    // * O[n], partial over n
            }
        }
        // reduce over the 16 N-lanes of each half-wave
        #pragma unroll
        for (int msk = 1; msk <= 8; msk <<= 1)
            #pragma unroll
            for (int r = 0; r < 8; ++r)
                p[r] += __shfl_xor(p[r], msk, 32);
        if (laneN == 0) {
            #pragma unroll
            for (int r = 0; r < 8; ++r)
                alpha_s[m * 16 + hi * 8 + r] = p[r];
        }
    }
    __syncthreads();

    // ---------- Phase 2: masked softmax over S ----------
    if (tid < SPAD) {
        float a = alpha_s[tid];
        if (tid < SEQ) a -= mask[(size_t)b * SEQ + tid] * 1e10f;
        else           a = NEG_BIG;
        alpha_s[tid] = a;
    }
    __syncthreads();

    float v = (tid < SPAD) ? alpha_s[tid] : NEG_BIG;
    float mx = v;
    #pragma unroll
    for (int msk = 16; msk >= 1; msk >>= 1)
        mx = fmaxf(mx, __shfl_xor(mx, msk, 32));
    if (lane == 0) red_s[wave] = mx;
    __syncthreads();
    mx = red_s[0];
    #pragma unroll
    for (int i = 1; i < 8; ++i) mx = fmaxf(mx, red_s[i]);

    float e = (tid < SPAD) ? __expf(v - mx) : 0.f;
    float sm = e;
    #pragma unroll
    for (int msk = 16; msk >= 1; msk >>= 1)
        sm += __shfl_xor(sm, msk, 32);
    if (lane == 0) red_s[8 + wave] = sm;
    __syncthreads();
    float tot = red_s[8];
    #pragma unroll
    for (int i = 1; i < 8; ++i) tot += red_s[8 + i];
    float inv = 1.f / tot;
    if (tid < SPAD) alpha_s[tid] = e * inv;
    __syncthreads();

    // ---------- Phase 3: pooling out[b,d] = sum_s attn[s]*his[s,d] ----------
    const int d    = tid & 127;
    const int half = tid >> 7;
    float acc = 0.f;
    const int s0 = half * (SEQ / 2);
    #pragma unroll 4
    for (int s = s0; s < s0 + SEQ / 2; ++s)
        acc = __builtin_fmaf(alpha_s[s], his_s[s * DIM + d], acc);
    if (half == 0) pool_s[d] = acc;
    __syncthreads();
    if (half == 1) out[(size_t)b * DIM + d] = pool_s[d] + acc;
}

extern "C" void kernel_launch(void* const* d_in, const int* in_sizes, int n_in,
                              void* d_out, int out_size, void* d_ws, size_t ws_size,
                              hipStream_t stream) {
    const float* his    = (const float*)d_in[0];  // [B,S,D]
    const float* target = (const float*)d_in[1];  // [B,1,D]
    const float* mask   = (const float*)d_in[2];  // [B,S]
    const float* W      = (const float*)d_in[3];  // [3D,H]
    const float* Wb     = (const float*)d_in[4];  // [H]
    const float* O      = (const float*)d_in[5];  // [H,1]
    // d_in[6] = O_bias : constant shift pre-softmax, cancels in softmax
    float* out = (float*)d_out;                   // [B,D]
    (void)in_sizes; (void)n_in; (void)out_size; (void)d_ws; (void)ws_size;

    target_attn_kernel<<<BATCH, 256, 0, stream>>>(his, target, mask, W, Wb, O, out);
}
